// Encoder_50053548868064
// MI455X (gfx1250) — matrix operand order repaired and checked by reading, not compiled, not run on hardware
//
#include <hip/hip_runtime.h>
#include <hip/hip_bf16.h>

// ---------------------------------------------------------------------------
// Types for CDNA5 WMMA (wave32): D(16x16 f32) = A(16x32 bf16) * B(32x16 bf16) + C
// ---------------------------------------------------------------------------
typedef __bf16 bf16;
typedef bf16  v16bf __attribute__((ext_vector_type(16)));
typedef bf16  v8bf  __attribute__((ext_vector_type(8)));
typedef float v8f   __attribute__((ext_vector_type(8)));

#define HDIM 512
#define BATCH 64
#define TLEN 512
#define NGATE 1536   // 3*H

__device__ __forceinline__ v8f zero8() {
  v8f z = {0.f,0.f,0.f,0.f,0.f,0.f,0.f,0.f};
  return z;
}

__device__ __forceinline__ v8f wmma_bf16(v16bf a, v16bf b, v8f c) {
  // v_wmma_f32_16x16x32_bf16
  return __builtin_amdgcn_wmma_f32_16x16x32_bf16(
      /*neg_a=*/false, a, /*neg_b=*/false, b,
      /*c_mod=*/(short)0, c, /*reuse_a=*/false, /*reuse_b=*/false);
}

// A fragment (16x32 bf16, row-major source with leading dim `ld`):
//   lanes 0-15 : M=lane,   VGPR0-3 = K 0..7,  VGPR4-7 = K 16..23
//   lanes 16-31: M=lane-16,VGPR0-3 = K 8..15, VGPR4-7 = K 24..31
__device__ __forceinline__ v16bf load_frag_a(const bf16* __restrict__ base,
                                             int ld, int lane, int k0) {
  const int row  = lane & 15;
  const int koff = (lane >> 4) * 8;
  const v8bf lo = *(const v8bf*)(base + (size_t)row * ld + k0 + koff);
  const v8bf hi = *(const v8bf*)(base + (size_t)row * ld + k0 + 16 + koff);
  return __builtin_shufflevector(lo, hi, 0,1,2,3,4,5,6,7,8,9,10,11,12,13,14,15);
}

// B fragment (32x16): B[k][n] = W[n0 + n][k]  (W row-major [rows][512] bf16)
//   lane = N column (lane&15); lanes 0-15 hold K 0..7,16..23, lanes 16-31 hold K 8..15,24..31
//   => each lane reads two 16-byte chunks of one W row.
__device__ __forceinline__ v16bf load_frag_b(const bf16* __restrict__ w,
                                             int lane, int n0, int k0) {
  const int j    = n0 + (lane & 15);
  const int koff = (lane >> 4) * 8;
  const v8bf lo = *(const v8bf*)(w + (size_t)j * HDIM + k0 + koff);
  const v8bf hi = *(const v8bf*)(w + (size_t)j * HDIM + k0 + 16 + koff);
  return __builtin_shufflevector(lo, hi, 0,1,2,3,4,5,6,7,8,9,10,11,12,13,14,15);
}

__device__ __forceinline__ float fsigmoid(float x) {
  return 1.f / (1.f + __expf(-x));
}

// ---------------------------------------------------------------------------
// 1) fp32 -> bf16 convert (weights)
// ---------------------------------------------------------------------------
__global__ void f2bf_kernel(const float* __restrict__ src, bf16* __restrict__ dst, int n) {
  int i = blockIdx.x * 256 + threadIdx.x;
  if (i < n) dst[i] = (bf16)src[i];
}

// ---------------------------------------------------------------------------
// 2) Embedding gather (+ per-batch time reversal for backward dir) -> bf16
// ---------------------------------------------------------------------------
__global__ void embed_gather_kernel(const int* __restrict__ seq,        // [T][B]
                                    const int* __restrict__ lens,       // [B]
                                    const float* __restrict__ table,    // [V][H]
                                    bf16* __restrict__ embf,            // [T*B][H]
                                    bf16* __restrict__ embb) {          // [T*B][H]
  const int tb = blockIdx.x;           // 0 .. T*B-1
  const int t  = tb >> 6;
  const int b  = tb & 63;
  const int idx = seq[t * BATCH + b];
  int rt = lens[b] - 1 - t; if (rt < 0) rt = 0;
  const int idr = seq[rt * BATCH + b];
  const float* rf = table + (size_t)idx * HDIM;
  const float* rb = table + (size_t)idr * HDIM;
  bf16* of = embf + (size_t)tb * HDIM;
  bf16* ob = embb + (size_t)tb * HDIM;
  for (int k = threadIdx.x; k < HDIM; k += 256) {
    of[k] = (bf16)rf[k];
    ob[k] = (bf16)rb[k];
  }
}

// ---------------------------------------------------------------------------
// 3) Time-parallel input projection: gi = emb @ Wih^T + bih  (both directions)
//    M = T*B = 32768, N = 1536, K = 512. Each wave: 64(M) x 16(N), B reused.
//    k-loop: explicit ping-pong software pipeline, rolled (#pragma unroll 1).
// ---------------------------------------------------------------------------
__global__ __launch_bounds__(256)
void gi_gemm_kernel(const bf16* __restrict__ embf, const bf16* __restrict__ embb,
                    const bf16* __restrict__ wihf, const bf16* __restrict__ wihb,
                    const float* __restrict__ bihf, const float* __restrict__ bihb,
                    float* __restrict__ gif, float* __restrict__ gib) {
  const int tid  = threadIdx.x;
  const int lane = tid & 31;
  const int wave = tid >> 5;
  const int w    = blockIdx.x * 8 + wave;     // 0 .. 98303
  const int dir  = w / 49152;
  const int rem  = w % 49152;
  const int m0   = (rem / 96) * 64;
  const int n0   = (rem % 96) * 16;

  const bf16*  emb = dir ? embb : embf;
  const bf16*  wih = dir ? wihb : wihf;
  const float* bih = dir ? bihb : bihf;
  float*       gi  = dir ? gib  : gif;

  const bf16* arow0 = emb + (size_t)(m0 +  0) * HDIM;
  const bf16* arow1 = emb + (size_t)(m0 + 16) * HDIM;
  const bf16* arow2 = emb + (size_t)(m0 + 32) * HDIM;
  const bf16* arow3 = emb + (size_t)(m0 + 48) * HDIM;

  v8f acc0 = zero8(), acc1 = zero8(), acc2 = zero8(), acc3 = zero8();

  // prologue: k-block 0 -> P
  v16bf pb  = load_frag_b(wih, lane, n0, 0);
  v16bf pa0 = load_frag_a(arow0, HDIM, lane, 0);
  v16bf pa1 = load_frag_a(arow1, HDIM, lane, 0);
  v16bf pa2 = load_frag_a(arow2, HDIM, lane, 0);
  v16bf pa3 = load_frag_a(arow3, HDIM, lane, 0);
  v16bf qb, qa0, qa1, qa2, qa3;

#pragma unroll 1
  for (int blk = 0; blk < 14; blk += 2) {
    const int kA = (blk + 1) * 32;
    qb  = load_frag_b(wih, lane, n0, kA);
    qa0 = load_frag_a(arow0, HDIM, lane, kA);
    qa1 = load_frag_a(arow1, HDIM, lane, kA);
    qa2 = load_frag_a(arow2, HDIM, lane, kA);
    qa3 = load_frag_a(arow3, HDIM, lane, kA);
    acc0 = wmma_bf16(pa0, pb, acc0);
    acc1 = wmma_bf16(pa1, pb, acc1);
    acc2 = wmma_bf16(pa2, pb, acc2);
    acc3 = wmma_bf16(pa3, pb, acc3);
    const int kB = (blk + 2) * 32;
    pb  = load_frag_b(wih, lane, n0, kB);
    pa0 = load_frag_a(arow0, HDIM, lane, kB);
    pa1 = load_frag_a(arow1, HDIM, lane, kB);
    pa2 = load_frag_a(arow2, HDIM, lane, kB);
    pa3 = load_frag_a(arow3, HDIM, lane, kB);
    acc0 = wmma_bf16(qa0, qb, acc0);
    acc1 = wmma_bf16(qa1, qb, acc1);
    acc2 = wmma_bf16(qa2, qb, acc2);
    acc3 = wmma_bf16(qa3, qb, acc3);
  }
  // epilogue: P holds block 14; fetch block 15 into Q, consume both
  {
    const int kL = 15 * 32;
    qb  = load_frag_b(wih, lane, n0, kL);
    qa0 = load_frag_a(arow0, HDIM, lane, kL);
    qa1 = load_frag_a(arow1, HDIM, lane, kL);
    qa2 = load_frag_a(arow2, HDIM, lane, kL);
    qa3 = load_frag_a(arow3, HDIM, lane, kL);
    acc0 = wmma_bf16(pa0, pb, acc0);
    acc1 = wmma_bf16(pa1, pb, acc1);
    acc2 = wmma_bf16(pa2, pb, acc2);
    acc3 = wmma_bf16(pa3, pb, acc3);
    acc0 = wmma_bf16(qa0, qb, acc0);
    acc1 = wmma_bf16(qa1, qb, acc1);
    acc2 = wmma_bf16(qa2, qb, acc2);
    acc3 = wmma_bf16(qa3, qb, acc3);
  }

  const int j    = n0 + (lane & 15);
  const float bias = bih[j];
  const int mro  = 8 * (lane >> 4);
  v8f accs[4] = {acc0, acc1, acc2, acc3};
#pragma unroll
  for (int i = 0; i < 4; ++i) {
#pragma unroll
    for (int r = 0; r < 8; ++r) {
      gi[(size_t)(m0 + i * 16 + mro + r) * NGATE + j] = accs[i][r] + bias;
    }
  }
}

// ---------------------------------------------------------------------------
// 4/5) GRU recurrence. One block = 16 batch rows (M=16), 16 waves.
//      Per step: 16x1536x512 GEMM (h @ Whh^T) via WMMA; h bf16 in LDS (A op),
//      h fp32 persistent in registers in the C/D tile layout.
//      Wave w owns H-columns [32w, 32w+32): 2 slices x {r,z,n} gate tiles.
//      gi lines prefetched before the GEMM; k-loop is a rolled ping-pong
//      software pipeline.
// ---------------------------------------------------------------------------
__global__ __launch_bounds__(512)
void gru_recurrent_kernel(const bf16*  __restrict__ whh_bf,  // [1536][512] bf16
                          const float* __restrict__ gi,      // [T][64][1536]
                          const float* __restrict__ bhh,     // [1536]
                          const int*   __restrict__ lens,    // [64]
                          float* __restrict__ out,           // [T][64][512]
                          float* __restrict__ hid,           // [64][512] (dir slice)
                          int dir) {
  __shared__ bf16 lds_h[16 * HDIM];                          // 16 KB

  const int tid   = threadIdx.x;
  const int lane  = tid & 31;
  const int wave  = tid >> 5;              // 0..15
  const int b0    = blockIdx.x * 16;       // batch base for this block
  const int lhalf = lane >> 4;             // 0/1
  const int jcol  = lane & 15;             // N column within tile

  for (int i = tid; i < 16 * HDIM; i += 512) lds_h[i] = (bf16)0.f;

  int mylen[8];
#pragma unroll
  for (int r = 0; r < 8; ++r) mylen[r] = lens[b0 + 8 * lhalf + r];

  float hreg[2][8];
#pragma unroll
  for (int s = 0; s < 2; ++s)
#pragma unroll
    for (int r = 0; r < 8; ++r) hreg[s][r] = 0.f;

  float bh[2][3];
#pragma unroll
  for (int s = 0; s < 2; ++s) {
    const int n0 = (wave * 2 + s) * 16;
    bh[s][0] = bhh[n0 + jcol];
    bh[s][1] = bhh[HDIM + n0 + jcol];
    bh[s][2] = bhh[2 * HDIM + n0 + jcol];
  }

  __syncthreads();

  for (int t = 0; t < TLEN; ++t) {
    const float* gi_t = gi + (size_t)t * BATCH * NGATE;

    // Prefetch this step's gi cache lines (96 lines/wave = 3 x 32 lanes).
    // lane -> (batch row = lane&15, combo = p*2 + lane>>4 in 0..5).
#pragma unroll
    for (int p = 0; p < 3; ++p) {
      const int comb = p * 2 + lhalf;            // 0..5
      const int g    = comb >> 1;                // gate 0..2
      const int ss   = comb & 1;                 // slice 0..1
      const int col  = g * HDIM + (wave * 2 + ss) * 16;
      __builtin_prefetch(gi_t + (size_t)(b0 + (lane & 15)) * NGATE + col, 0, 0);
    }

#pragma unroll
    for (int s = 0; s < 2; ++s) {
      const int n0 = (wave * 2 + s) * 16;
      v8f accR = zero8(), accZ = zero8(), accN = zero8();

      // prologue: k-block 0 -> P
      v16bf pa  = load_frag_a(lds_h, HDIM, lane, 0);
      v16bf pbr = load_frag_b(whh_bf, lane, n0, 0);
      v16bf pbz = load_frag_b(whh_bf, lane, HDIM + n0, 0);
      v16bf pbn = load_frag_b(whh_bf, lane, 2 * HDIM + n0, 0);
      v16bf qa, qbr, qbz, qbn;

#pragma unroll 1
      for (int blk = 0; blk < 14; blk += 2) {
        const int kA = (blk + 1) * 32;
        qa  = load_frag_a(lds_h, HDIM, lane, kA);
        qbr = load_frag_b(whh_bf, lane, n0, kA);
        qbz = load_frag_b(whh_bf, lane, HDIM + n0, kA);
        qbn = load_frag_b(whh_bf, lane, 2 * HDIM + n0, kA);
        accR = wmma_bf16(pa, pbr, accR);
        accZ = wmma_bf16(pa, pbz, accZ);
        accN = wmma_bf16(pa, pbn, accN);
        const int kB = (blk + 2) * 32;
        pa  = load_frag_a(lds_h, HDIM, lane, kB);
        pbr = load_frag_b(whh_bf, lane, n0, kB);
        pbz = load_frag_b(whh_bf, lane, HDIM + n0, kB);
        pbn = load_frag_b(whh_bf, lane, 2 * HDIM + n0, kB);
        accR = wmma_bf16(qa, qbr, accR);
        accZ = wmma_bf16(qa, qbz, accZ);
        accN = wmma_bf16(qa, qbn, accN);
      }
      // epilogue: P holds block 14; fetch block 15 into Q, consume both
      {
        const int kL = 15 * 32;
        qa  = load_frag_a(lds_h, HDIM, lane, kL);
        qbr = load_frag_b(whh_bf, lane, n0, kL);
        qbz = load_frag_b(whh_bf, lane, HDIM + n0, kL);
        qbn = load_frag_b(whh_bf, lane, 2 * HDIM + n0, kL);
        accR = wmma_bf16(pa, pbr, accR);
        accZ = wmma_bf16(pa, pbz, accZ);
        accN = wmma_bf16(pa, pbn, accN);
        accR = wmma_bf16(qa, qbr, accR);
        accZ = wmma_bf16(qa, qbz, accZ);
        accN = wmma_bf16(qa, qbn, accN);
      }

      const int j = n0 + jcol;
#pragma unroll
      for (int r = 0; r < 8; ++r) {
        const int bl = 8 * lhalf + r;                       // local batch row
        const float* gib = gi_t + (size_t)(b0 + bl) * NGATE;
        const float gir = gib[j];
        const float giz = gib[HDIM + j];
        const float gin = gib[2 * HDIM + j];
        const float ghr = accR[r] + bh[s][0];
        const float ghz = accZ[r] + bh[s][1];
        const float ghn = accN[r] + bh[s][2];
        const float rg = fsigmoid(gir + ghr);
        const float zg = fsigmoid(giz + ghz);
        const float ng = tanhf(gin + rg * ghn);
        const float hn = (1.f - zg) * ng + zg * hreg[s][r];
        const bool  m  = (t < mylen[r]);
        hreg[s][r] = m ? hn : hreg[s][r];
        if (dir == 0) {
          out[((size_t)t * BATCH + (b0 + bl)) * HDIM + j] = m ? hn : 0.f;
        } else if (m) {
          const int pos = mylen[r] - 1 - t;                 // un-reverse
          const size_t oi = ((size_t)pos * BATCH + (b0 + bl)) * HDIM + j;
          out[oi] += hn;                                    // out_f already there
        }
      }
    }
    __syncthreads();   // all waves done reading old h (A fragments)
#pragma unroll
    for (int s = 0; s < 2; ++s) {
      const int n0 = (wave * 2 + s) * 16;
#pragma unroll
      for (int r = 0; r < 8; ++r) {
        const int bl = 8 * lhalf + r;
        lds_h[bl * HDIM + n0 + jcol] = (bf16)hreg[s][r];
      }
    }
    __syncthreads();   // new h visible for next step
  }

  // final hidden state
#pragma unroll
  for (int s = 0; s < 2; ++s) {
    const int n0 = (wave * 2 + s) * 16;
#pragma unroll
    for (int r = 0; r < 8; ++r) {
      const int bl = 8 * lhalf + r;
      hid[(size_t)(b0 + bl) * HDIM + n0 + jcol] = hreg[s][r];
    }
  }
}

// ---------------------------------------------------------------------------
// Host-side launcher
// ---------------------------------------------------------------------------
extern "C" void kernel_launch(void* const* d_in, const int* in_sizes, int n_in,
                              void* d_out, int out_size, void* d_ws, size_t ws_size,
                              hipStream_t stream) {
  (void)in_sizes; (void)n_in; (void)out_size; (void)ws_size;

  const int*   seq   = (const int*)  d_in[0];   // [T][B]
  const int*   lens  = (const int*)  d_in[1];   // [B]
  const float* table = (const float*)d_in[2];   // [V][H]
  const float* Wih_f = (const float*)d_in[3];
  const float* Whh_f = (const float*)d_in[4];
  const float* bih_f = (const float*)d_in[5];
  const float* bhh_f = (const float*)d_in[6];
  const float* Wih_b = (const float*)d_in[7];
  const float* Whh_b = (const float*)d_in[8];
  const float* bih_b = (const float*)d_in[9];
  const float* bhh_b = (const float*)d_in[10];

  float* out = (float*)d_out;                          // [T][B][H] then [2][B][H]
  float* hid_f = out + (size_t)TLEN * BATCH * HDIM;
  float* hid_b = hid_f + (size_t)BATCH * HDIM;

  // workspace layout (256B aligned chunks)
  char* ws = (char*)d_ws;
  size_t off = 0;
  auto take = [&](size_t bytes) { char* p = ws + off; off += (bytes + 255) & ~(size_t)255; return (void*)p; };
  const size_t TB = (size_t)TLEN * BATCH;
  bf16* embf    = (bf16*)take(TB * HDIM * sizeof(bf16));
  bf16* embb    = (bf16*)take(TB * HDIM * sizeof(bf16));
  bf16* wihf_bf = (bf16*)take((size_t)NGATE * HDIM * sizeof(bf16));
  bf16* whhf_bf = (bf16*)take((size_t)NGATE * HDIM * sizeof(bf16));
  bf16* wihb_bf = (bf16*)take((size_t)NGATE * HDIM * sizeof(bf16));
  bf16* whhb_bf = (bf16*)take((size_t)NGATE * HDIM * sizeof(bf16));
  float* gif    = (float*)take(TB * NGATE * sizeof(float));
  float* gib    = (float*)take(TB * NGATE * sizeof(float));

  // 1) weight conversion
  const int NW = NGATE * HDIM;                 // 786432
  const int gc = (NW + 255) / 256;
  f2bf_kernel<<<gc, 256, 0, stream>>>(Wih_f, wihf_bf, NW);
  f2bf_kernel<<<gc, 256, 0, stream>>>(Whh_f, whhf_bf, NW);
  f2bf_kernel<<<gc, 256, 0, stream>>>(Wih_b, wihb_bf, NW);
  f2bf_kernel<<<gc, 256, 0, stream>>>(Whh_b, whhb_bf, NW);

  // 2) embedding gather + reversal
  embed_gather_kernel<<<TLEN * BATCH, 256, 0, stream>>>(seq, lens, table, embf, embb);

  // 3) gate preactivations (both directions): 2*512*96 waves / 8 per block
  gi_gemm_kernel<<<12288, 256, 0, stream>>>(embf, embb, wihf_bf, wihb_bf,
                                            bih_f, bih_b, gif, gib);

  // 4) forward recurrence (writes out_f incl. masked zeros, hid_f)
  gru_recurrent_kernel<<<4, 512, 0, stream>>>(whhf_bf, gif, bhh_f, lens,
                                              out, hid_f, 0);
  // 5) backward recurrence (accumulates into out, writes hid_b)
  gru_recurrent_kernel<<<4, 512, 0, stream>>>(whhb_bf, gib, bhh_b, lens,
                                              out, hid_b, 1);
}
